// Model_10814727651341
// MI455X (gfx1250) — compile-verified, matrix operand order checked by reading, not run on hardware
//
#include <hip/hip_runtime.h>
#include <hip/hip_bf16.h>
#include <cstdint>

// ---------------------------------------------------------------------------
// GAT stack for MI455X (gfx1250, wave32).
//   * Dense projections h = x @ W on v_wmma_f32_16x16x32_f16, K fully
//     unrolled (4 WMMAs for K=128). A-tile staged to LDS once per block via
//     global_load_async_to_lds_b128 (ASYNCcnt), read back as ds_load_b128.
//     B pre-packed per-lane so each fragment is two global_load_b128.
//   * Edge softmax/aggregation is the memory-bound part (~0.6 GB/layer,
//     L2-resident: h is 25.6MB << 192MB L2); f32 + atomics.
//   * SimpleAttLayer in the reference is softmax over a length-1 axis ->
//     exact identity -> skipped.
// ---------------------------------------------------------------------------

#define GAT_N 50000
#define GAT_E 600000
#define GAT_H 128
#define GAT_L 2

typedef __attribute__((ext_vector_type(16))) _Float16 v16h;
typedef __attribute__((ext_vector_type(8)))  float    v8f;

union Frag16 { v16h v; float4 q[2]; };

// ---------------- helpers ----------------

__device__ __forceinline__ unsigned f2ord(float f) {
    unsigned u = __float_as_uint(f);
    return (u & 0x80000000u) ? ~u : (u | 0x80000000u);
}
__device__ __forceinline__ float ord2f(unsigned u) {
    return __uint_as_float((u & 0x80000000u) ? (u & 0x7FFFFFFFu) : ~u);
}
__device__ __forceinline__ float selu_f(float x) {
    const float scale = 1.0507009873554805f;
    const float alpha = 1.6732632423543772f;
    return scale * (x > 0.f ? x : alpha * (__expf(x) - 1.f));
}

// Async global->LDS copy of 16 bytes per lane (CDNA5 async-memory path,
// tracked on ASYNCcnt). lds_off is the LDS byte offset (low 32 bits of a
// flat shared-space pointer per the aperture mapping rules).
__device__ __forceinline__ void async_b128_to_lds(unsigned lds_off, const void* gptr) {
    asm volatile("global_load_async_to_lds_b128 %0, %1, off"
                 :: "v"(lds_off), "v"(gptr) : "memory");
}
__device__ __forceinline__ void wait_asynccnt0() {
    asm volatile("s_wait_asynccnt 0" ::: "memory");
}

// ---------------- conversion: f32 -> f16 with zero padding ----------------

__global__ void cvt_pad_kernel(const float* __restrict__ src, _Float16* __restrict__ dst,
                               int srcRows, int srcCols, int dstRows, int dstCols) {
    int i = blockIdx.x * blockDim.x + threadIdx.x;
    int total = dstRows * dstCols;
    if (i >= total) return;
    int r = i / dstCols, c = i - r * dstCols;
    float v = (r < srcRows && c < srcCols) ? src[(size_t)r * srcCols + c] : 0.f;
    dst[i] = (_Float16)v;
}

// ---------------- weight pre-pack into WMMA B-fragment order ----------------
// Packed index: (((kt*8 + ct)*32 + lane)*16 + e), where lane's fragment
// element e maps to k = kt*32 + (e&8?16:0) + (lane>>4)*8 + (e&7),
// col = ct*16 + (lane&15).  Rows k >= srcK are zero padding.

__global__ void pack_b_kernel(const float* __restrict__ W, _Float16* __restrict__ Bp,
                              int srcK, int K) {
    int i = blockIdx.x * blockDim.x + threadIdx.x;
    int total = K * GAT_H;
    if (i >= total) return;
    int e    = i & 15;
    int lane = (i >> 4) & 31;
    int ct   = (i >> 9) & 7;
    int kt   = i >> 12;
    int kh   = (lane >> 4) * 8;
    int koff = ((e & 8) ? 16 : 0) + kh + (e & 7);
    int k    = kt * 32 + koff;
    int col  = ct * 16 + (lane & 15);
    float v  = (k < srcK) ? W[(size_t)k * GAT_H + col] : 0.f;
    Bp[i] = (_Float16)v;
}

// ---------------- WMMA GEMM: C[N,128] = A[N,K] @ B[K,128] ----------------
// One wave per 16x16 tile of C; 8 waves/block cover 16 rows x 128 cols.
// A tile (16xK f16) staged to LDS via async loads; B pre-packed (Bp).
// N multiple of 16, K in {32,128}. f16 operands, f32 accumulation.

template <int K>
__global__ __launch_bounds__(256)
void wmma_gemm_kernel(const _Float16* __restrict__ A, const _Float16* __restrict__ Bp,
                      float* __restrict__ C) {
    __shared__ _Float16 tileA[16 * K];

    const int t       = threadIdx.x;
    const int wave    = t >> 5;             // 0..7 -> column tile
    const int lane    = t & 31;
    const int half_id = lane >> 4;          // 0 or 1
    const int mn      = lane & 15;          // row (for A) / col (for B,C)
    const int row0    = blockIdx.x << 4;
    const int col0    = wave << 4;
    const int kh      = half_id * 8;        // K offset contributed by lane half

    // ---- stage A tile: 16*K*2 bytes, 16B per participating thread ----
    const int tileBytes = 16 * K * 2;
    if (t * 16 < tileBytes) {
        unsigned lds_off = (unsigned)(uintptr_t)(&tileA[0]) + t * 16;
        const char* g = (const char*)(A + (size_t)row0 * K) + t * 16;
        async_b128_to_lds(lds_off, g);
    }
    wait_asynccnt0();
    __syncthreads();

    v8f acc = {};
#pragma unroll
    for (int kt = 0; kt < K / 32; ++kt) {
        const int k0 = kt * 32;
        Frag16 fa, fb;
        // A fragment: two contiguous 8-half runs per lane (ds_load_b128 x2)
        fa.q[0] = *(const float4*)(tileA + mn * K + k0 + kh);
        fa.q[1] = *(const float4*)(tileA + mn * K + k0 + 16 + kh);
        // B fragment: pre-packed, contiguous 32B per lane (global b128 x2)
        const _Float16* bp = Bp + (((size_t)kt * 8 + wave) * 32 + lane) * 16;
        fb.q[0] = *(const float4*)(bp);
        fb.q[1] = *(const float4*)(bp + 8);
        // (neg_a, A, neg_b, B, c_mod, C, reuse_a, reuse_b)
        acc = __builtin_amdgcn_wmma_f32_16x16x32_f16(false, fa.v, false, fb.v,
                                                     (short)0, acc, false, false);
    }
#pragma unroll
    for (int r = 0; r < 8; ++r) {
        // f32 C/D layout: VGPR r holds M = r + 8*half_id, N = lane&15
        C[(size_t)(row0 + r + kh) * GAT_H + (col0 + mn)] = acc[r];
    }
}

// ---------------- f1 = h@a1, f2 = h@a2 (wave per node) ----------------

__global__ void fvec_kernel(const float* __restrict__ h, const float* __restrict__ a1,
                            const float* __restrict__ a2, float* __restrict__ f1,
                            float* __restrict__ f2, int n) {
    int node = blockIdx.x * (blockDim.x >> 5) + (threadIdx.x >> 5);
    int lane = threadIdx.x & 31;
    if (node >= n) return;
    float s1 = 0.f, s2 = 0.f;
    const float* row = h + (size_t)node * GAT_H;
#pragma unroll
    for (int j = 0; j < GAT_H; j += 32) {
        float hv = row[j + lane];
        s1 += hv * a1[j + lane];
        s2 += hv * a2[j + lane];
    }
#pragma unroll
    for (int off = 16; off; off >>= 1) {
        s1 += __shfl_xor(s1, off, 32);
        s2 += __shfl_xor(s2, off, 32);
    }
    if (lane == 0) { f1[node] = s1; f2[node] = s2; }
}

// ---------------- edge passes ----------------

__global__ void edge_logits_kernel(const float* __restrict__ f1, const float* __restrict__ f2,
                                   const int* __restrict__ src, const int* __restrict__ dst,
                                   float* __restrict__ logits, unsigned* __restrict__ mu, int nE) {
    int e = blockIdx.x * blockDim.x + threadIdx.x;
    if (e >= nE) return;
    int s = src[e], d = dst[e];
    float l = f1[s] + f2[d];
    l = (l > 0.f) ? l : 0.2f * l;           // leaky_relu(0.2)
    logits[e] = l;
    atomicMax(&mu[d], f2ord(l));            // segment max (ordered-uint encoding)
}

__global__ void edge_exp_kernel(float* __restrict__ logits, const unsigned* __restrict__ mu,
                                const int* __restrict__ dst, float* __restrict__ ssum, int nE) {
    int e = blockIdx.x * blockDim.x + threadIdx.x;
    if (e >= nE) return;
    int d = dst[e];
    float ev = __expf(logits[e] - ord2f(mu[d]));
    logits[e] = ev;                          // reuse buffer: now holds exp values
    atomicAdd(&ssum[d], ev);
}

// 32 lanes per edge, float4 per lane: agg[dst] += coef * h[src]
__global__ void edge_agg_kernel(const float* __restrict__ evals, const float* __restrict__ ssum,
                                const int* __restrict__ src, const int* __restrict__ dst,
                                const float* __restrict__ h, float* __restrict__ agg, int nE) {
    int gid = blockIdx.x * blockDim.x + threadIdx.x;
    int e = gid >> 5;
    if (e >= nE) return;
    int q = gid & 31;
    int s = src[e], d = dst[e];
    float coef = evals[e] / (ssum[d] + 1e-9f);
    const float4 hv = reinterpret_cast<const float4*>(h + (size_t)s * GAT_H)[q];
    float* dp = agg + (size_t)d * GAT_H + q * 4;
    atomicAdd(dp + 0, coef * hv.x);
    atomicAdd(dp + 1, coef * hv.y);
    atomicAdd(dp + 2, coef * hv.z);
    atomicAdd(dp + 3, coef * hv.w);
}

// ---------------- finalize ----------------

__global__ void finalize_layer0_kernel(const float* __restrict__ agg, const float* __restrict__ b,
                                       const float* __restrict__ res, float* __restrict__ fin, int total) {
    int i = blockIdx.x * blockDim.x + threadIdx.x;
    if (i >= total) return;
    fin[i] = selu_f(agg[i] + b[i & (GAT_H - 1)] + res[i]);
}

__global__ void finalize_resid_kernel(const float* __restrict__ agg, const float* __restrict__ b,
                                      float* __restrict__ fin, int total) {
    int i = blockIdx.x * blockDim.x + threadIdx.x;
    if (i >= total) return;
    float x = fin[i];
    fin[i] = x + selu_f(agg[i] + b[i & (GAT_H - 1)] + x);  // final += selu(agg + b + final)
}

// ---------------- head: leaky_relu(final[N-1] @ Wfc + bfc) ----------------

__global__ void head_kernel(const float* __restrict__ fin, const float* __restrict__ Wfc,
                            const float* __restrict__ bfc, float* __restrict__ out, int lastRow) {
    __shared__ float part[4];
    int t = threadIdx.x;                                   // 128 threads
    float v = fin[(size_t)lastRow * GAT_H + t] * Wfc[t];
#pragma unroll
    for (int off = 16; off; off >>= 1) v += __shfl_xor(v, off, 32);
    if ((t & 31) == 0) part[t >> 5] = v;
    __syncthreads();
    if (t == 0) {
        float y = part[0] + part[1] + part[2] + part[3] + bfc[0];
        out[0] = (y > 0.f) ? y : 0.01f * y;
    }
}

// ---------------- host orchestration ----------------

static inline unsigned cdiv_u(unsigned a, unsigned b) { return (a + b - 1) / b; }

extern "C" void kernel_launch(void* const* d_in, const int* in_sizes, int n_in,
                              void* d_out, int out_size, void* d_ws, size_t ws_size,
                              hipStream_t stream) {
    (void)in_sizes; (void)n_in; (void)out_size; (void)ws_size;

    const float* x    = (const float*)d_in[0];
    const int*   ei   = (const int*)d_in[1];        // [2,E]; row0=src, row1=dst
    const float* W0   = (const float*)d_in[2];
    const float* a1_0 = (const float*)d_in[3];
    const float* a2_0 = (const float*)d_in[4];
    const float* b0   = (const float*)d_in[5];
    const float* Wr0  = (const float*)d_in[6];
    // d_in[7..9]: Wa0/ba0/ua0 — SimpleAttLayer is identity (softmax over len-1 axis)
    const float* Wf   = (const float*)d_in[10];     // [L,128,128]
    const float* a1f  = (const float*)d_in[11];     // [L,128]
    const float* a2f  = (const float*)d_in[12];
    const float* bf   = (const float*)d_in[13];
    // d_in[14..16]: Waf/baf/uaf — identity, skipped
    const float* Wfc  = (const float*)d_in[17];
    const float* bfc  = (const float*)d_in[18];
    float* out = (float*)d_out;

    const int* srcI = ei;
    const int* dstI = ei + GAT_E;

    // --- workspace carve-up (all offsets 256B aligned) ---
    char*  ws  = (char*)d_ws;
    size_t off = 0;
    auto carve = [&](size_t bytes) -> char* {
        char* p = ws + off;
        off += (bytes + 255) & ~(size_t)255;
        return p;
    };
    _Float16* X16   = (_Float16*)carve((size_t)GAT_N * 32 * 2);
    _Float16* W0P   = (_Float16*)carve(32 * GAT_H * 2);        // packed, K=32
    _Float16* WR0P  = (_Float16*)carve(32 * GAT_H * 2);        // packed, K=32
    _Float16* WFP   = (_Float16*)carve(GAT_H * GAT_H * 2);     // packed, K=128
    _Float16* FIN16 = (_Float16*)carve((size_t)GAT_N * GAT_H * 2);
    float*    Hbuf  = (float*)carve((size_t)GAT_N * GAT_H * 4);
    float*    Res0  = (float*)carve((size_t)GAT_N * GAT_H * 4);
    float*    Final = (float*)carve((size_t)GAT_N * GAT_H * 4);
    float*    Agg   = (float*)carve((size_t)GAT_N * GAT_H * 4);
    float*    F1    = (float*)carve((size_t)GAT_N * 4);
    float*    F2    = (float*)carve((size_t)GAT_N * 4);
    unsigned* MU    = (unsigned*)carve((size_t)GAT_N * 4);
    float*    Ssum  = (float*)carve((size_t)GAT_N * 4);
    float*    Elog  = (float*)carve((size_t)GAT_E * 4);

    const unsigned NB256_NH = cdiv_u((unsigned)GAT_N * GAT_H, 256);
    const unsigned NB256_E  = cdiv_u(GAT_E, 256);
    const unsigned NB_AGG   = cdiv_u((unsigned)GAT_E * 32, 256);
    const unsigned NB_FVEC  = cdiv_u(GAT_N, 8);           // 8 waves/block, wave per node
    const unsigned NB_GEMM  = GAT_N / 16;                  // N % 16 == 0

    // ---- input conversions / weight packing ----
    cvt_pad_kernel<<<cdiv_u(GAT_N * 32, 256), 256, 0, stream>>>(x, X16, GAT_N, 7, GAT_N, 32);
    pack_b_kernel<<<cdiv_u(32 * GAT_H, 256), 256, 0, stream>>>(W0,  W0P,  7, 32);
    pack_b_kernel<<<cdiv_u(32 * GAT_H, 256), 256, 0, stream>>>(Wr0, WR0P, 7, 32);

    // ---- layer 0: h = x@W0 (K=32 padded), res = x@Wr0 ----
    wmma_gemm_kernel<32><<<NB_GEMM, 256, 0, stream>>>(X16, W0P,  Hbuf);
    wmma_gemm_kernel<32><<<NB_GEMM, 256, 0, stream>>>(X16, WR0P, Res0);
    fvec_kernel<<<NB_FVEC, 256, 0, stream>>>(Hbuf, a1_0, a2_0, F1, F2, GAT_N);

    hipMemsetAsync(MU, 0, (size_t)GAT_N * 4, stream);      // ord-encoding 0 == -inf sentinel
    hipMemsetAsync(Ssum, 0, (size_t)GAT_N * 4, stream);
    hipMemsetAsync(Agg, 0, (size_t)GAT_N * GAT_H * 4, stream);
    edge_logits_kernel<<<NB256_E, 256, 0, stream>>>(F1, F2, srcI, dstI, Elog, MU, GAT_E);
    edge_exp_kernel<<<NB256_E, 256, 0, stream>>>(Elog, MU, dstI, Ssum, GAT_E);
    edge_agg_kernel<<<NB_AGG, 256, 0, stream>>>(Elog, Ssum, srcI, dstI, Hbuf, Agg, GAT_E);
    finalize_layer0_kernel<<<NB256_NH, 256, 0, stream>>>(Agg, b0, Res0, Final, GAT_N * GAT_H);

    // ---- layers 1..L: final += selu(GAT(final)) ----
    for (int i = 0; i < GAT_L; ++i) {
        cvt_pad_kernel<<<NB256_NH, 256, 0, stream>>>(Final, FIN16, GAT_N, GAT_H, GAT_N, GAT_H);
        pack_b_kernel<<<cdiv_u(GAT_H * GAT_H, 256), 256, 0, stream>>>(
            Wf + (size_t)i * GAT_H * GAT_H, WFP, GAT_H, GAT_H);
        wmma_gemm_kernel<128><<<NB_GEMM, 256, 0, stream>>>(FIN16, WFP, Hbuf);
        fvec_kernel<<<NB_FVEC, 256, 0, stream>>>(Hbuf, a1f + i * GAT_H, a2f + i * GAT_H, F1, F2, GAT_N);

        hipMemsetAsync(MU, 0, (size_t)GAT_N * 4, stream);
        hipMemsetAsync(Ssum, 0, (size_t)GAT_N * 4, stream);
        hipMemsetAsync(Agg, 0, (size_t)GAT_N * GAT_H * 4, stream);
        edge_logits_kernel<<<NB256_E, 256, 0, stream>>>(F1, F2, srcI, dstI, Elog, MU, GAT_E);
        edge_exp_kernel<<<NB256_E, 256, 0, stream>>>(Elog, MU, dstI, Ssum, GAT_E);
        edge_agg_kernel<<<NB_AGG, 256, 0, stream>>>(Elog, Ssum, srcI, dstI, Hbuf, Agg, GAT_E);
        finalize_resid_kernel<<<NB256_NH, 256, 0, stream>>>(Agg, bf + i * GAT_H, Final, GAT_N * GAT_H);
    }

    // ---- head on last node ----
    head_kernel<<<1, 128, 0, stream>>>(Final, Wfc, bfc, out, GAT_N - 1);
}
